// CrystalGraphConvNet_4312147165769
// MI455X (gfx1250) — matrix operand order repaired and checked by reading, not compiled, not run on hardware
//
#include <hip/hip_runtime.h>
#include <hip/hip_bf16.h>

// ---------------------------------------------------------------------------
// CGCNN forward on MI455X (gfx1250, wave32, WMMA + TDM)
// ---------------------------------------------------------------------------
#define NN    100000
#define MM    12
#define ORIGF 92
#define NBRF  41
#define NBRP  48       // edge features padded to 48 halves (16B-vector friendly)
#define FF    64
#define HH    128
#define N0C   2000
#define TWO_F 128
#define KPAD  192      // conv K (2F+NBR=169) padded to 6x32
#define KEMB  96       // embedding K (92) padded to 3x32
#define LDA_E 104      // LDS row stride for embed A-tile
#define LDA_S 72       // LDS row stride (halves) for self/neighbor tiles
#define EDGE_TILE (16 * MM * NBRP)   // 9216 halves: contiguous edge block per 16-atom tile

#if defined(__AMDGCN__) && defined(__has_builtin)
#  if __has_builtin(__builtin_amdgcn_tensor_load_to_lds) && __has_builtin(__builtin_amdgcn_s_wait_tensorcnt)
#    define HAVE_TDM 1
#  endif
#endif
#ifndef HAVE_TDM
#  define HAVE_TDM 0
#endif

typedef __attribute__((ext_vector_type(16))) _Float16 v16h;
typedef __attribute__((ext_vector_type(8)))  _Float16 v8h;
typedef __attribute__((ext_vector_type(8)))  float    v8f;
typedef __attribute__((ext_vector_type(4)))  unsigned int u32x4;
typedef __attribute__((ext_vector_type(4)))  int i32x4;
typedef __attribute__((ext_vector_type(8)))  int i32x8;

__device__ __forceinline__ float sp_f(float x)  { return fmaxf(x, 0.f) + log1pf(expf(-fabsf(x))); }
__device__ __forceinline__ float sig_f(float x) { return 1.f / (1.f + expf(-x)); }

// A fragment (16x32 f16) from row-major LDS tile (ISA §7.12.2 layout).
__device__ __forceinline__ v16h load_a_frag_lds(const _Float16* tile, int ldk, int kbase, int lane) {
  int r  = lane & 15;
  int kh = (lane >> 4) << 3;          // 0 or 8
  const _Float16* p = tile + r * ldk + kbase + kh;
  v8h lo = *(const v8h*)(p);          // ds_load_b128
  v8h hi = *(const v8h*)(p + 16);
  v16h a;
#pragma unroll
  for (int e = 0; e < 8; ++e) { a[e] = lo[e]; a[e + 8] = hi[e]; }
  return a;
}

// B fragment (32K x 16 cols f16) from global k-major weights W[K][ld].
__device__ __forceinline__ v16h load_b_frag(const _Float16* W, int ld, int kbase, int col0, int lane) {
  int n  = col0 + (lane & 15);
  int kb = kbase + ((lane >> 4) << 3);
  v16h b;
#pragma unroll
  for (int e = 0; e < 16; ++e) b[e] = W[(kb + e + ((e >> 3) << 3)) * ld + n];
  return b;
}

__device__ __forceinline__ v8f wmma_step(v16h a, v16h b, v8f c) {
  return __builtin_amdgcn_wmma_f32_16x16x32_f16(false, a, false, b, (short)0, c, false, false);
}

// ---------------------------------------------------------------------------
// TDM: 2D tile load global -> LDS. rows x rowElems (2-byte elements),
// global row stride = strideElems, LDS destination contiguous.
// D# per CDNA5 ISA (group0 128b, group1 256b).
// ---------------------------------------------------------------------------
#if HAVE_TDM
__device__ __forceinline__ void tdm_load_2d(_Float16* lds_dst, const _Float16* gsrc,
                                            unsigned rowElems, unsigned rows, unsigned strideElems) {
  unsigned long long ga = (unsigned long long)(size_t)gsrc;
  unsigned ldsOff = (unsigned)(size_t)(void*)lds_dst;   // generic addr low 32b == LDS byte offset
  u32x4 g0;
  g0.x = 1u;                                              // count=1, is_restore=0, no gather
  g0.y = ldsOff;                                          // lds_addr
  g0.z = (unsigned)ga;                                    // global_addr[31:0]
  g0.w = (unsigned)((ga >> 32) & 0x01FFFFFFull) | (2u << 30);  // global_addr[56:32], type=2
  i32x8 g1;
  g1[0] = (int)(1u << 16);                                // workgroup_mask=0, data_size=1 (2B)
  g1[1] = (int)((rowElems & 0xFFFFu) << 16);              // tensor_dim0[15:0]
  g1[2] = (int)(((rowElems >> 16) & 0xFFFFu) | ((rows & 0xFFFFu) << 16)); // dim0 hi | tensor_dim1 lo
  g1[3] = (int)(((rows >> 16) & 0xFFFFu) | ((rowElems & 0xFFFFu) << 16)); // dim1 hi | tile_dim0
  g1[4] = (int)(rows & 0xFFFFu);                          // tile_dim1 | tile_dim2=0
  g1[5] = (int)strideElems;                               // tensor_dim0_stride[31:0]
  g1[6] = 0;                                              // stride hi | tensor_dim1_stride lo
  g1[7] = 0;
  i32x4 gz = {0, 0, 0, 0};
#if __clang_major__ >= 23
  i32x8 gz8 = {0, 0, 0, 0, 0, 0, 0, 0};
  __builtin_amdgcn_tensor_load_to_lds(g0, g1, gz, gz, gz8, 0);
#else
  __builtin_amdgcn_tensor_load_to_lds(g0, g1, gz, gz, 0);
#endif
}
#endif

// ---------------------------------------------------------------------------
// Kernel: precision conversion + zero-init of accumulation buffers
// ---------------------------------------------------------------------------
__global__ void k_prep(const float* __restrict__ nbr_fea, _Float16* __restrict__ nbr16p,
                       const float* __restrict__ W_emb, _Float16* __restrict__ wemb16,
                       const float* __restrict__ cW, _Float16* __restrict__ cw16,
                       const float* __restrict__ rW, _Float16* __restrict__ rw16,
                       float* __restrict__ pooled, float* __restrict__ cnt,
                       float* __restrict__ bn1sum, float* __restrict__ bn1sq,
                       float* __restrict__ bn2sum, float* __restrict__ bn2sq) {
  long stride = (long)gridDim.x * blockDim.x;
  long t0 = (long)blockIdx.x * blockDim.x + threadIdx.x;
  for (long i = t0; i < (long)NN * MM * NBRP; i += stride) {
    long row = i / NBRP; int j = (int)(i - row * NBRP);
    nbr16p[i] = (j < NBRF) ? (_Float16)nbr_fea[row * NBRF + j] : (_Float16)0.f;
  }
  for (long i = t0; i < (long)KEMB * FF; i += stride) {
    int k = (int)(i / FF), j = (int)(i % FF);
    wemb16[i] = (k < ORIGF) ? (_Float16)W_emb[k * FF + j] : (_Float16)0.f;
  }
  for (long i = t0; i < (long)KPAD * TWO_F; i += stride) {
    int k = (int)(i / TWO_F), j = (int)(i % TWO_F);
    cw16[i] = (k < 2 * FF + NBRF) ? (_Float16)cW[k * TWO_F + j] : (_Float16)0.f;
    rw16[i] = (k < 2 * FF + NBRF) ? (_Float16)rW[k * TWO_F + j] : (_Float16)0.f;
  }
  for (long i = t0; i < (long)N0C * FF; i += stride) pooled[i] = 0.f;
  for (long i = t0; i < N0C; i += stride) cnt[i] = 0.f;
  for (long i = t0; i < TWO_F; i += stride) { bn1sum[i] = 0.f; bn1sq[i] = 0.f; }
  for (long i = t0; i < FF; i += stride)    { bn2sum[i] = 0.f; bn2sq[i] = 0.f; }
}

// ---------------------------------------------------------------------------
// Kernel: embedding GEMM  a = atom_fea @ W_emb + b_emb   (WMMA f16->f32)
// ---------------------------------------------------------------------------
__global__ void k_embed(const float* __restrict__ atom_fea,
                        const _Float16* __restrict__ wemb16,
                        const float* __restrict__ b_emb,
                        float* __restrict__ a32, _Float16* __restrict__ a16) {
  __shared__ __attribute__((aligned(16))) _Float16 At[32 * LDA_E];
  int tid = threadIdx.x;
  int wave = tid >> 5, lane = tid & 31;
  int rt = wave >> 2;
  int colbase = (wave & 3) << 4;
  v16h B0 = load_b_frag(wemb16, FF, 0,  colbase, lane);
  v16h B1 = load_b_frag(wemb16, FF, 32, colbase, lane);
  v16h B2 = load_b_frag(wemb16, FF, 64, colbase, lane);
  float bj = b_emb[colbase + (lane & 15)];
  int ntiles = NN / 32;  // 3125
  for (int t = blockIdx.x; t < ntiles; t += gridDim.x) {
    int n0 = t * 32;
    __syncthreads();
    for (int e = tid; e < 32 * LDA_E; e += 256) {
      int r = e / LDA_E, k = e - r * LDA_E;
      At[e] = (k < ORIGF) ? (_Float16)atom_fea[(n0 + r) * ORIGF + k] : (_Float16)0.f;
    }
    __syncthreads();
    const _Float16* tile = At + rt * 16 * LDA_E;
    v8f c = {0.f, 0.f, 0.f, 0.f, 0.f, 0.f, 0.f, 0.f};
    c = wmma_step(load_a_frag_lds(tile, LDA_E, 0,  lane), B0, c);
    c = wmma_step(load_a_frag_lds(tile, LDA_E, 32, lane), B1, c);
    c = wmma_step(load_a_frag_lds(tile, LDA_E, 64, lane), B2, c);
    int col = colbase + (lane & 15);
    int rbase = n0 + rt * 16 + ((lane >> 4) << 3);
#pragma unroll
    for (int v = 0; v < 8; ++v) {
      float val = c[v] + bj;
      a32[(rbase + v) * FF + col] = val;
      a16[(rbase + v) * FF + col] = (_Float16)val;
    }
  }
}

// ---------------------------------------------------------------------------
// Conv-layer GEMM. STATS: per-channel sum/sumsq of gated = tot@W + b.
// MAIN: BN1 + sigmoid*softplus, reduce over M in LDS, write summed, BN2 stats.
// Block: 256 threads / 8 waves; 16-atom tile; wave w owns output cols [16w,16w+16).
// Self feats staged once/tile; neighbor gather as one b128/thread per m;
// edge feats DMA'd per tile by the Tensor Data Mover (double buffered).
// ---------------------------------------------------------------------------
template <bool STATS>
__global__ void k_conv(const _Float16* __restrict__ ain16,
                       const int* __restrict__ idx,
                       const _Float16* __restrict__ nbr16p,  // NN*MM rows x 48 halves
                       const _Float16* __restrict__ w16,     // KPAD x 128 (k-major)
                       const float* __restrict__ bias,
                       const float* __restrict__ bnscale,
                       const float* __restrict__ bnshift,
                       float* __restrict__ bn1sum, float* __restrict__ bn1sq,
                       float* __restrict__ summed,           // N x 64 (MAIN)
                       float* __restrict__ bn2sum, float* __restrict__ bn2sq) {
  __shared__ __attribute__((aligned(16))) _Float16 Aself[16 * LDA_S];
  __shared__ __attribute__((aligned(16))) _Float16 Anbr [16 * LDA_S];
  __shared__ __attribute__((aligned(16))) _Float16 Aed[2][EDGE_TILE + 16]; // +16 slack (x0-weight reads)
  __shared__ int   nbidx[16];
  __shared__ float sigb[16 * FF];
  __shared__ float spb [16 * FF];
  __shared__ float sumb[16 * FF];

  int tid = threadIdx.x;
  int wave = tid >> 5, lane = tid & 31;
  int colbase = wave << 4;
  int col = colbase + (lane & 15);

  v16h B[6];
#pragma unroll
  for (int kc = 0; kc < 6; ++kc) B[kc] = load_b_frag(w16, TWO_F, kc * 32, colbase, lane);
  float bj = bias[col];
  float sc = 0.f, sh = 0.f;
  if (!STATS) { sc = bnscale[col]; sh = bnshift[col]; }

  float ssum = 0.f, ssq = 0.f;   // BN1 stats (STATS)
  float s2 = 0.f, s2q = 0.f;     // BN2 stats (MAIN), channel tid&63
  int f0 = tid & 63;

  const int ntiles = NN / 16;    // 6250
  int par = 0;
#if HAVE_TDM
  if (tid < 32 && blockIdx.x < ntiles)
    tdm_load_2d(Aed[0], nbr16p + (size_t)blockIdx.x * EDGE_TILE, MM * NBRP, 16, MM * NBRP);
#endif
  for (int t = blockIdx.x; t < ntiles; t += gridDim.x) {
    int n0 = t * 16;
    const _Float16* edge = Aed[par];
#if HAVE_TDM
    if (tid < 32) __builtin_amdgcn_s_wait_tensorcnt(0);   // current tile's edge DMA done
    __syncthreads();
    int tn = t + gridDim.x;
    if (tid < 32 && tn < ntiles)                          // overlap next tile's DMA with compute
      tdm_load_2d(Aed[par ^ 1], nbr16p + (size_t)tn * EDGE_TILE, MM * NBRP, 16, MM * NBRP);
#else
    __syncthreads();
    for (int e = tid; e < EDGE_TILE / 8; e += 256)
      ((v8h*)Aed[par])[e] = ((const v8h*)(nbr16p + (size_t)t * EDGE_TILE))[e];
#endif
    // stage self features once per tile: 16 rows x 64 halves (128 b128 moves)
    for (int e = tid; e < 128; e += 256) {
      int r = e >> 3, c8 = (e & 7) << 3;
      *(v8h*)(Aself + r * LDA_S + c8) = *(const v8h*)(ain16 + (size_t)(n0 + r) * FF + c8);
    }
    if (tid == 0 && t + (int)gridDim.x < ntiles)
      __builtin_prefetch(ain16 + (size_t)(t + gridDim.x) * 16 * FF, 0, 0);  // global_prefetch_b8
    if (!STATS) {
      for (int e = tid; e < 16 * FF; e += 256) sumb[e] = 0.f;
    }
    for (int m = 0; m < MM; ++m) {
      __syncthreads();
      if (tid < 16) nbidx[tid] = idx[(n0 + tid) * MM + m];
      __syncthreads();
      // neighbor gather: 16 rows x 64 halves, one b128 per thread (tid<128)
      if (tid < 128) {
        int r = tid >> 3, c8 = (tid & 7) << 3;
        *(v8h*)(Anbr + r * LDA_S + c8) = *(const v8h*)(ain16 + (size_t)nbidx[r] * FF + c8);
      }
      __syncthreads();
      v8f c = {0.f, 0.f, 0.f, 0.f, 0.f, 0.f, 0.f, 0.f};
      c = wmma_step(load_a_frag_lds(Aself, LDA_S, 0,  lane), B[0], c);
      c = wmma_step(load_a_frag_lds(Aself, LDA_S, 32, lane), B[1], c);
      c = wmma_step(load_a_frag_lds(Anbr,  LDA_S, 0,  lane), B[2], c);
      c = wmma_step(load_a_frag_lds(Anbr,  LDA_S, 32, lane), B[3], c);
      c = wmma_step(load_a_frag_lds(edge + m * NBRP, MM * NBRP, 0,  lane), B[4], c);
      c = wmma_step(load_a_frag_lds(edge + m * NBRP, MM * NBRP, 32, lane), B[5], c);

      if (STATS) {
#pragma unroll
        for (int v = 0; v < 8; ++v) { float g = c[v] + bj; ssum += g; ssq += g * g; }
      } else {
        int rbase = (lane >> 4) << 3;
        int cc = col & 63;
#pragma unroll
        for (int v = 0; v < 8; ++v) {
          float g = (c[v] + bj) * sc + sh;                       // BN1
          if (wave < 4) sigb[(rbase + v) * FF + cc] = sig_f(g);  // filter half
          else          spb [(rbase + v) * FF + cc] = sp_f(g);   // core half
        }
        __syncthreads();
        for (int e = tid; e < 16 * FF; e += 256) sumb[e] += sigb[e] * spb[e];
      }
    }
    if (!STATS) {
      __syncthreads();
      for (int e = tid; e < 16 * FF; e += 256) {
        int r = e >> 6;
        float v = sumb[e];
        summed[(n0 + r) * FF + (e & 63)] = v;
        s2 += v; s2q += v * v;   // e%64 == tid%64 always
      }
    }
    __syncthreads();   // all reads of this tile's LDS done before restaging
    par ^= 1;
  }
  if (STATS) {
    atomicAdd(&bn1sum[col], ssum);
    atomicAdd(&bn1sq[col],  ssq);
  } else {
    atomicAdd(&bn2sum[f0], s2);
    atomicAdd(&bn2sq[f0],  s2q);
  }
}

// ---------------------------------------------------------------------------
// Finalize BN params, then re-zero the stats accumulators for the next layer.
// ---------------------------------------------------------------------------
__global__ void k_fin_bn(const float* __restrict__ g, const float* __restrict__ bt,
                         float* __restrict__ sum, float* __restrict__ sq,
                         float* __restrict__ scale, float* __restrict__ shift, float count) {
  int c = threadIdx.x;
  float mu  = sum[c] / count;
  float var = sq[c] / count - mu * mu;
  float s   = g[c] * rsqrtf(var + 1e-5f);
  scale[c] = s;
  shift[c] = bt[c] - mu * s;
  sum[c] = 0.f; sq[c] = 0.f;
}

// ---------------------------------------------------------------------------
// Post-conv elementwise: v = softplus(aprev + BN2(summed)); layer-2 extra
// residual: v = softplus(v + aprev).
// ---------------------------------------------------------------------------
__global__ void k_postconv(const float* __restrict__ aprev, const float* __restrict__ summed,
                           const float* __restrict__ scale, const float* __restrict__ shift,
                           float* __restrict__ out32, _Float16* __restrict__ out16, int extra) {
  int i = blockIdx.x * blockDim.x + threadIdx.x;
  if (i >= NN * FF) return;
  int f = i & 63;
  float s = summed[i] * scale[f] + shift[f];
  float v = sp_f(aprev[i] + s);
  if (extra) v = sp_f(v + aprev[i]);
  out32[i] = v;
  if (out16) out16[i] = (_Float16)v;
}

// ---------------------------------------------------------------------------
// Segment-sum pooling by crystal_id (L2 f32 atomics)
// ---------------------------------------------------------------------------
__global__ void k_pool(const float* __restrict__ a2, const int* __restrict__ cid,
                       float* __restrict__ pooled, float* __restrict__ cnt) {
  int i = blockIdx.x * blockDim.x + threadIdx.x;
  if (i >= NN * FF) return;
  int n = i >> 6, f = i & 63;
  int c = cid[n];
  atomicAdd(&pooled[c * FF + f], a2[i]);
  if (f == 0) atomicAdd(&cnt[c], 1.f);
}

// ---------------------------------------------------------------------------
// Head: one block per crystal: mean -> softplus -> fc(64x128) -> softplus -> out
// ---------------------------------------------------------------------------
__global__ void k_head(const float* __restrict__ pooled, const float* __restrict__ cnt,
                       const float* __restrict__ W_fc, const float* __restrict__ b_fc,
                       const float* __restrict__ W_out, const float* __restrict__ b_out,
                       float* __restrict__ out) {
  __shared__ float cr[FF];
  __shared__ float red[HH];
  int b = blockIdx.x, tid = threadIdx.x;
  if (tid < FF) {
    float c = pooled[b * FF + tid] / fmaxf(cnt[b], 1.f);
    cr[tid] = sp_f(c);
  }
  __syncthreads();
  float acc = b_fc[tid];
  for (int k = 0; k < FF; ++k) acc += cr[k] * W_fc[k * HH + tid];
  red[tid] = sp_f(acc) * W_out[tid];
  __syncthreads();
  for (int s = HH / 2; s > 0; s >>= 1) {
    if (tid < s) red[tid] += red[tid + s];
    __syncthreads();
  }
  if (tid == 0) out[b] = red[0] + b_out[0];
}

// ---------------------------------------------------------------------------
// Host launcher
// ---------------------------------------------------------------------------
static size_t aloc(size_t& off, size_t bytes) {
  off = (off + 255) & ~(size_t)255;
  size_t cur = off;
  off += bytes;
  return cur;
}

extern "C" void kernel_launch(void* const* d_in, const int* in_sizes, int n_in,
                              void* d_out, int out_size, void* d_ws, size_t ws_size,
                              hipStream_t stream) {
  const float* atom_fea = (const float*)d_in[0];
  const float* nbr_fea  = (const float*)d_in[1];
  const int*   nbr_idx  = (const int*)d_in[2];
  const int*   cryst_id = (const int*)d_in[3];
  const float* W_emb = (const float*)d_in[4];
  const float* b_emb = (const float*)d_in[5];
  const float* cW  = (const float*)d_in[6];
  const float* cb  = (const float*)d_in[7];
  const float* cg1 = (const float*)d_in[8];
  const float* cbt1= (const float*)d_in[9];
  const float* cg2 = (const float*)d_in[10];
  const float* cbt2= (const float*)d_in[11];
  const float* rW  = (const float*)d_in[12];
  const float* rb  = (const float*)d_in[13];
  const float* rg1 = (const float*)d_in[14];
  const float* rbt1= (const float*)d_in[15];
  const float* rg2 = (const float*)d_in[16];
  const float* rbt2= (const float*)d_in[17];
  const float* W_fc  = (const float*)d_in[18];
  const float* b_fc  = (const float*)d_in[19];
  const float* W_out = (const float*)d_in[20];
  const float* b_out = (const float*)d_in[21];
  float* out = (float*)d_out;

  char* ws = (char*)d_ws;
  size_t off = 0;
  _Float16* nbr16p = (_Float16*)(ws + aloc(off, (size_t)NN * MM * NBRP * 2));
  _Float16* wemb16 = (_Float16*)(ws + aloc(off, (size_t)KEMB * FF * 2));
  _Float16* cw16   = (_Float16*)(ws + aloc(off, (size_t)KPAD * TWO_F * 2));
  _Float16* rw16   = (_Float16*)(ws + aloc(off, (size_t)KPAD * TWO_F * 2));
  float*    a32    = (float*)(ws + aloc(off, (size_t)NN * FF * 4));
  _Float16* a16    = (_Float16*)(ws + aloc(off, (size_t)NN * FF * 2));
  float*    a1_32  = (float*)(ws + aloc(off, (size_t)NN * FF * 4));
  _Float16* a1_16  = (_Float16*)(ws + aloc(off, (size_t)NN * FF * 2));
  float*    a2_32  = (float*)(ws + aloc(off, (size_t)NN * FF * 4));
  float*    summed = (float*)(ws + aloc(off, (size_t)NN * FF * 4));
  float*    bn1sum = (float*)(ws + aloc(off, TWO_F * 4));
  float*    bn1sq  = (float*)(ws + aloc(off, TWO_F * 4));
  float*    bn1sc  = (float*)(ws + aloc(off, TWO_F * 4));
  float*    bn1sh  = (float*)(ws + aloc(off, TWO_F * 4));
  float*    bn2sum = (float*)(ws + aloc(off, FF * 4));
  float*    bn2sq  = (float*)(ws + aloc(off, FF * 4));
  float*    bn2sc  = (float*)(ws + aloc(off, FF * 4));
  float*    bn2sh  = (float*)(ws + aloc(off, FF * 4));
  float*    pooled = (float*)(ws + aloc(off, (size_t)N0C * FF * 4));
  float*    cntb   = (float*)(ws + aloc(off, (size_t)N0C * 4));
  (void)ws_size; (void)in_sizes; (void)n_in; (void)out_size;

  const int EW = NN * FF;
  const int ewBlocks = (EW + 255) / 256;

  k_prep<<<2048, 256, 0, stream>>>(nbr_fea, nbr16p, W_emb, wemb16, cW, cw16, rW, rw16,
                                   pooled, cntb, bn1sum, bn1sq, bn2sum, bn2sq);
  k_embed<<<512, 256, 0, stream>>>(atom_fea, wemb16, b_emb, a32, a16);

  // ---- conv layer 1 ----
  k_conv<true><<<640, 256, 0, stream>>>(a16, nbr_idx, nbr16p, cw16, cb,
                                        nullptr, nullptr, bn1sum, bn1sq,
                                        nullptr, nullptr, nullptr);
  k_fin_bn<<<1, TWO_F, 0, stream>>>(cg1, cbt1, bn1sum, bn1sq, bn1sc, bn1sh,
                                    (float)((size_t)NN * MM));
  k_conv<false><<<640, 256, 0, stream>>>(a16, nbr_idx, nbr16p, cw16, cb,
                                         bn1sc, bn1sh, bn1sum, bn1sq,
                                         summed, bn2sum, bn2sq);
  k_fin_bn<<<1, FF, 0, stream>>>(cg2, cbt2, bn2sum, bn2sq, bn2sc, bn2sh, (float)NN);
  k_postconv<<<ewBlocks, 256, 0, stream>>>(a32, summed, bn2sc, bn2sh, a1_32, a1_16, 0);

  // ---- conv layer 2 (residual) ----
  k_conv<true><<<640, 256, 0, stream>>>(a1_16, nbr_idx, nbr16p, rw16, rb,
                                        nullptr, nullptr, bn1sum, bn1sq,
                                        nullptr, nullptr, nullptr);
  k_fin_bn<<<1, TWO_F, 0, stream>>>(rg1, rbt1, bn1sum, bn1sq, bn1sc, bn1sh,
                                    (float)((size_t)NN * MM));
  k_conv<false><<<640, 256, 0, stream>>>(a1_16, nbr_idx, nbr16p, rw16, rb,
                                         bn1sc, bn1sh, bn1sum, bn1sq,
                                         summed, bn2sum, bn2sq);
  k_fin_bn<<<1, FF, 0, stream>>>(rg2, rbt2, bn2sum, bn2sq, bn2sc, bn2sh, (float)NN);
  k_postconv<<<ewBlocks, 256, 0, stream>>>(a1_32, summed, bn2sc, bn2sh, a2_32, nullptr, 1);

  // ---- pool + head ----
  k_pool<<<ewBlocks, 256, 0, stream>>>(a2_32, cryst_id, pooled, cntb);
  k_head<<<N0C, HH, 0, stream>>>(pooled, cntb, W_fc, b_fc, W_out, b_out, out);
}
